// PersistentEntityState_54631984005732
// MI455X (gfx1250) — matrix operand order repaired and checked by reading, not compiled, not run on hardware
//
#include <hip/hip_runtime.h>
#include <stdint.h>

#define N_ENTITY 100000
#define H_DIM    512
#define N_ROWS   262144

typedef __attribute__((ext_vector_type(16))) __bf16 v16bf;
typedef __attribute__((ext_vector_type(8)))  __bf16 bf16x8;
typedef __attribute__((ext_vector_type(8)))  float  v8f;

__device__ __forceinline__ unsigned short f2bf(float f) {
    unsigned u = __float_as_uint(f);
    u += 0x7FFFu + ((u >> 16) & 1u);   // round-to-nearest-even
    return (unsigned short)(u >> 16);
}

// sigmoid via v_exp_f32 + v_rcp_f32 (both TRANS ops, co-execute with VALU)
__device__ __forceinline__ float sigmoid_fast(float x) {
    return __builtin_amdgcn_rcpf(1.f + __expf(-x));
}

// Guaranteed native fp32 atomic add to global memory (no CAS loop).
__device__ __forceinline__ void gfadd(float* p, float v) {
    asm volatile("global_atomic_add_f32 %0, %1, off"
                 :: "v"((unsigned long long)(uintptr_t)p), "v"(v)
                 : "memory");
}

// ---------------- W f32 -> bf16 conversion ----------------
__global__ void cvt_w_kernel(const float* __restrict__ W,
                             unsigned short* __restrict__ Wbf, int n) {
    int i = blockIdx.x * blockDim.x + threadIdx.x;
    if (i < n) Wbf[i] = f2bf(W[i]);
}

// ---------------- fused gate path: gather + LN + bf16 WMMA GEMM +
//                  sigmoid blend + atomic segment scatter ----------------
__global__ __launch_bounds__(256)
void fuse_gate_kernel(const float* __restrict__ emb,
                      const float* __restrict__ slow,
                      const unsigned short* __restrict__ Wbf,  // (512,1024) bf16
                      const float* __restrict__ bias,
                      const int*   __restrict__ ids,
                      float* __restrict__ outp,
                      float* __restrict__ ctx_sum,    // == new_fast region, pre-zeroed
                      unsigned int* __restrict__ counts) {
    extern __shared__ char smem[];
    unsigned short* Abf = (unsigned short*)smem;                       // 16 x 1024 bf16 (32KB)
    float* embL = (float*)(smem + 16 * 1024 * 2);                      // 16 x 512 f32  (32KB)
    float* hsL  = (float*)(smem + 16 * 1024 * 2 + 16 * 512 * 4);       // 16 x 512 f32  (32KB)
    int*   sId  = (int*)(smem + 16 * 1024 * 2 + 2 * 16 * 512 * 4);     // [16]
    int*   sAct = sId + 16;                                            // [16]

    const int tid = threadIdx.x;
    const int rowBase = blockIdx.x * 16;

    // ---- Phase A: gather slow_state, layernorm, stage gate_in as bf16 ----
    {
        const int r = tid >> 4;      // row 0..15
        const int c = tid & 15;      // 16 threads per row (32 elems each)
        const int id = ids[rowBase + r];
        const float* srow = slow + (size_t)id * H_DIM;
        const float* erow = emb + (size_t)(rowBase + r) * H_DIM;
        float st[32];
        float sum = 0.f, sq = 0.f, sab = 0.f;
#pragma unroll
        for (int t = 0; t < 32; ++t) {
            float v = srow[c + (t << 4)];
            st[t] = v; sum += v; sq += v * v; sab += fabsf(v);
        }
#pragma unroll
        for (int m = 1; m < 16; m <<= 1) {   // reduce across the 16 lanes of this row
            sum += __shfl_xor(sum, m, 16);
            sq  += __shfl_xor(sq,  m, 16);
            sab += __shfl_xor(sab, m, 16);
        }
        float mu   = sum * (1.f / 512.f);
        float var  = sq * (1.f / 512.f) - mu * mu;
        float rstd = rsqrtf(var + 1e-5f);
        if (c == 0) { sId[r] = id; sAct[r] = (sab > 1e-6f) ? 1 : 0; }
#pragma unroll
        for (int t = 0; t < 32; ++t) {
            int j = c + (t << 4);
            float e = erow[j];
            embL[r * 512 + j]      = e;
            Abf[r * 1024 + j]      = f2bf(e);
            float h = (st[t] - mu) * rstd;
            hsL[r * 512 + j]        = h;
            Abf[r * 1024 + 512 + j] = f2bf(h);
        }
    }
    __syncthreads();

    if (tid < 16) atomicAdd(&counts[sId[tid]], 1u);

    // ---- Phase B: 16x512 = 8 waves x 4 N-tiles, K=1024 in 32 steps ----
    const int l    = tid & 31;
    const int w    = tid >> 5;
    const int hi   = l >> 4;      // 0 or 1 (lane half)
    const int lrow = l & 15;

    v8f acc[4] = {};
    const int aBase = lrow * 1024;
    for (int ks = 0; ks < 32; ++ks) {
        const int kb = ks * 32;
        // A fragment (16x32 bf16): K = kb + hi*16 + {0..15} (contiguous, matches B)
        const int ab = aBase + kb + hi * 16;
        bf16x8 alo = *(const bf16x8*)(Abf + ab);
        bf16x8 ahi = *(const bf16x8*)(Abf + ab + 8);
        v16bf a = __builtin_shufflevector(alo, ahi,
                    0, 1, 2, 3, 4, 5, 6, 7, 8, 9, 10, 11, 12, 13, 14, 15);
        const int kb2 = kb + hi * 16;
#pragma unroll
        for (int i = 0; i < 4; ++i) {
            // B fragment (32x16 bf16): col = lrow, K = kb2 + e (contiguous in W row)
            const int n = (w * 4 + i) * 16 + lrow;
            const unsigned short* wp = Wbf + (size_t)n * 1024 + kb2;
            bf16x8 b0 = *(const bf16x8*)wp;
            bf16x8 b1 = *(const bf16x8*)(wp + 8);
            v16bf bb = __builtin_shufflevector(b0, b1,
                         0, 1, 2, 3, 4, 5, 6, 7, 8, 9, 10, 11, 12, 13, 14, 15);
            acc[i] = __builtin_amdgcn_wmma_f32_16x16x32_bf16(
                         false, a, false, bb, (short)0, acc[i], false, false);
        }
    }

    // ---- Epilogue: bias + sigmoid + blend + store + atomic scatter ----
#pragma unroll
    for (int i = 0; i < 4; ++i) {
        const int n = (w * 4 + i) * 16 + lrow;
        const float bv = bias[n];
#pragma unroll
        for (int v = 0; v < 8; ++v) {
            const int m = v + hi * 8;              // C/D layout: row = vgpr + 8*(lane>=16)
            float z  = sigmoid_fast(acc[i][v] + bv);
            float e  = embL[m * 512 + n];
            float h  = hsL[m * 512 + n];
            float fused = z * e + (1.f - z) * h;
            float o = sAct[m] ? fused : e;
            outp[(size_t)(rowBase + m) * H_DIM + n] = o;
            gfadd(ctx_sum + (size_t)sId[m] * H_DIM + n, o);
        }
    }
}

// ---------------- finalize: mean -> EMA -> norm -> gated slow update ----------------
__global__ __launch_bounds__(128)
void finalize_kernel(const float* __restrict__ fast,
                     const float* __restrict__ slow,
                     const unsigned int* __restrict__ counts,
                     float* __restrict__ nf,     // in: ctx_sum, out: new_fast
                     float* __restrict__ ns) {   // out: new_slow
    __shared__ float red[4];
    const int e   = blockIdx.x;
    const int tid = threadIdx.x;
    const unsigned int c = counts[e];
    const bool touched = c > 0;
    const float inv = 1.f / fmaxf((float)c, 1.f);
    const size_t base = (size_t)e * H_DIM;

    float nfv[4], dv[4], sv[4];
    float ds2 = 0.f;
#pragma unroll
    for (int t = 0; t < 4; ++t) {
        const int j = tid + t * 128;
        float mean = nf[base + j] * inv;
        float f    = fast[base + j];
        float nfx  = touched ? (0.5f * f + 0.5f * mean) : f;
        float sl   = slow[base + j];
        float d    = nfx - sl;
        nfv[t] = nfx; dv[t] = d; sv[t] = sl;
        ds2 += d * d;
    }
#pragma unroll
    for (int m = 16; m >= 1; m >>= 1) ds2 += __shfl_xor(ds2, m, 32);
    if ((tid & 31) == 0) red[tid >> 5] = ds2;
    __syncthreads();
    const float delta = sqrtf(red[0] + red[1] + red[2] + red[3]);
    const float gate  = sigmoid_fast(5.f * (delta - 0.5f));
#pragma unroll
    for (int t = 0; t < 4; ++t) {
        const int j = tid + t * 128;
        nf[base + j] = nfv[t];
        ns[base + j] = touched ? (sv[t] + gate * dv[t]) : sv[t];
    }
}

extern "C" void kernel_launch(void* const* d_in, const int* in_sizes, int n_in,
                              void* d_out, int out_size, void* d_ws, size_t ws_size,
                              hipStream_t stream) {
    const float* emb  = (const float*)d_in[0];
    const float* slow = (const float*)d_in[1];
    const float* fast = (const float*)d_in[2];
    const float* W    = (const float*)d_in[3];
    const float* bias = (const float*)d_in[4];
    const int*   ids  = (const int*)d_in[5];

    float* outp = (float*)d_out;
    float* nf   = outp + (size_t)N_ROWS * H_DIM;
    float* ns   = nf + (size_t)N_ENTITY * H_DIM;

    unsigned short* Wbf = (unsigned short*)d_ws;                             // 1 MB
    unsigned int* counts =
        (unsigned int*)((char*)d_ws + (size_t)H_DIM * 2 * H_DIM * 2);        // 400 KB

    hipMemsetAsync(nf, 0, (size_t)N_ENTITY * H_DIM * sizeof(float), stream);
    hipMemsetAsync(counts, 0, (size_t)N_ENTITY * sizeof(unsigned int), stream);

    const int wElems = H_DIM * 2 * H_DIM;
    cvt_w_kernel<<<(wElems + 255) / 256, 256, 0, stream>>>(W, Wbf, wElems);

    const size_t smem = 16 * 1024 * 2 + 2 * 16 * 512 * 4 + 2 * 16 * 4;       // ~96.1 KB (<320KB/WGP)
    fuse_gate_kernel<<<N_ROWS / 16, 256, smem, stream>>>(emb, slow, Wbf, bias, ids,
                                                         outp, nf, counts);

    finalize_kernel<<<N_ENTITY, 128, 0, stream>>>(fast, slow, counts, nf, ns);
}
